// VectorQuantizer_29652454212407
// MI455X (gfx1250) — compile-verified, matrix operand order checked by reading, not run on hardware
//
#include <hip/hip_runtime.h>
#include <hip/hip_bf16.h>

#define D_MODEL   256
#define NUM_EMB   2048
#define N_ROWS    32768          // 8*16*16*16
#define KCHUNKS   8              // D_MODEL / 32
#define NTILES    (NUM_EMB / 16) // 128
#define COMMIT_BETA 0.25f

typedef __attribute__((ext_vector_type(16))) __bf16 v16bf;
typedef __attribute__((ext_vector_type(8)))  __bf16 v8bf;
typedef __attribute__((ext_vector_type(8)))  float  v8f;

// round-to-nearest-even fp32 -> bf16 (bit manipulation: no reliance on cvt codegen)
static __device__ __forceinline__ __bf16 f2bf(float f) {
    union { float f; unsigned u; } x; x.f = f;
    unsigned r = x.u + 0x7FFFu + ((x.u >> 16) & 1u);
    unsigned short h = (unsigned short)(r >> 16);
    union { unsigned short s; __bf16 b; } y; y.s = h;
    return y.b;
}

// ---------------------------------------------------------------------------
// Kernel 1: codebook fp32 -> bf16, plus ||e_k||^2 per row.
// One block (256 threads) per codebook row.
// ---------------------------------------------------------------------------
__global__ __launch_bounds__(256) void vq_prep_kernel(
    const float* __restrict__ emb, __bf16* __restrict__ emb_bf,
    float* __restrict__ e_norms)
{
    const int row = blockIdx.x;
    const int t   = threadIdx.x;
    const float v = emb[(size_t)row * D_MODEL + t];
    emb_bf[(size_t)row * D_MODEL + t] = f2bf(v);

    __shared__ float red[256];
    red[t] = v * v;
    __syncthreads();
    #pragma unroll
    for (int s = 128; s > 0; s >>= 1) {
        if (t < s) red[t] += red[t + s];
        __syncthreads();
    }
    if (t == 0) e_norms[row] = red[0];
}

// ---------------------------------------------------------------------------
// Kernel 2: fused distance-matmul + argmin via v_wmma_f32_16x16x32_bf16.
// One wave32 per 16-row tile of z; scans all 2048 codes in 16-col tiles.
// ---------------------------------------------------------------------------
__global__ __launch_bounds__(256) void vq_argmin_kernel(
    const float*  __restrict__ z,
    const __bf16* __restrict__ emb_bf,
    const float*  __restrict__ e_norms,
    int*          __restrict__ idx_ws,
    float*        __restrict__ idx_out)
{
    const int lane    = threadIdx.x & 31;
    const int wave    = threadIdx.x >> 5;
    const int rowBase = (blockIdx.x * 8 + wave) * 16;
    const int nrole   = lane & 15;   // row (A) / col (B,C) role within half-wave
    const int hi      = lane >> 4;   // which K-subset this lane carries

    // --- Build bf16 A fragments for the 16x256 z tile, kept in registers. ---
    // 16-bit A 16x32 layout: lane m (lanes 0-15) holds K = {0..7, 16..23},
    // lane m+16 holds K = {8..15, 24..31}; 2 packed per VGPR in K order.
    v16bf a[KCHUNKS];
    const float* zrow = z + (size_t)(rowBase + nrole) * D_MODEL;
    #pragma unroll
    for (int kc = 0; kc < KCHUNKS; ++kc) {
        const int d0 = kc * 32 + hi * 8;
        const float4 f0 = *reinterpret_cast<const float4*>(zrow + d0);
        const float4 f1 = *reinterpret_cast<const float4*>(zrow + d0 + 4);
        const float4 f2 = *reinterpret_cast<const float4*>(zrow + d0 + 16);
        const float4 f3 = *reinterpret_cast<const float4*>(zrow + d0 + 20);
        a[kc][0]  = f2bf(f0.x); a[kc][1]  = f2bf(f0.y);
        a[kc][2]  = f2bf(f0.z); a[kc][3]  = f2bf(f0.w);
        a[kc][4]  = f2bf(f1.x); a[kc][5]  = f2bf(f1.y);
        a[kc][6]  = f2bf(f1.z); a[kc][7]  = f2bf(f1.w);
        a[kc][8]  = f2bf(f2.x); a[kc][9]  = f2bf(f2.y);
        a[kc][10] = f2bf(f2.z); a[kc][11] = f2bf(f2.w);
        a[kc][12] = f2bf(f3.x); a[kc][13] = f2bf(f3.y);
        a[kc][14] = f2bf(f3.z); a[kc][15] = f2bf(f3.w);
    }

    float runMin[8];
    int   runIdx[8];
    #pragma unroll
    for (int v = 0; v < 8; ++v) { runMin[v] = 3.4e38f; runIdx[v] = 0; }

    for (int tile = 0; tile < NTILES; ++tile) {
        const int nBase = tile * 16;
        // B = E^T : per-lane data of B mirrors A's per-row pattern, sourced
        // from bf16 codebook row (nBase + nrole).
        const __bf16* erow = emb_bf + (size_t)(nBase + nrole) * D_MODEL;
        if (tile + 1 < NTILES) {
            __builtin_prefetch(emb_bf + (size_t)(nBase + 16 + nrole) * D_MODEL, 0, 1);
        }
        v8f c = {};
        #pragma unroll
        for (int kc = 0; kc < KCHUNKS; ++kc) {
            const int d0 = kc * 32 + hi * 8;
            const v8bf blo = *reinterpret_cast<const v8bf*>(erow + d0);
            const v8bf bhi = *reinterpret_cast<const v8bf*>(erow + d0 + 16);
            v16bf b;
            #pragma unroll
            for (int i = 0; i < 8; ++i) { b[i] = blo[i]; b[8 + i] = bhi[i]; }
            // (neg_a, A, neg_b, B, c_mod, C, reuse_a, reuse_b)
            c = __builtin_amdgcn_wmma_f32_16x16x32_bf16(
                    false, a[kc], false, b, (short)0, c, false, false);
        }
        // C/D layout: VGPR v of lane L -> row (v + 8*hi), col (L & 15).
        const int   myCol = nBase + nrole;
        const float en    = e_norms[myCol];
        #pragma unroll
        for (int v = 0; v < 8; ++v) {
            const float dist = en - 2.0f * c[v];      // ||z||^2 omitted (row-const)
            if (dist < runMin[v]) { runMin[v] = dist; runIdx[v] = myCol; }
        }
    }

    // Per-row argmin: reduce over the 16 column-lanes of each half-wave.
    #pragma unroll
    for (int v = 0; v < 8; ++v) {
        #pragma unroll
        for (int mask = 1; mask < 16; mask <<= 1) {
            const float om = __shfl_xor(runMin[v], mask, 32);
            const int   oi = __shfl_xor(runIdx[v], mask, 32);
            if (om < runMin[v] || (om == runMin[v] && oi < runIdx[v])) {
                runMin[v] = om; runIdx[v] = oi;
            }
        }
        if (nrole == 0) {
            const int row = rowBase + v + 8 * hi;
            idx_ws[row]  = runIdx[v];
            idx_out[row] = (float)runIdx[v];
        }
    }
}

// ---------------------------------------------------------------------------
// Kernel 3: z_q gather (fp32 codebook) + per-row sum((z_q - z)^2).
// One block (256 threads == D_MODEL) per latent row. Deterministic.
// ---------------------------------------------------------------------------
__global__ __launch_bounds__(256) void vq_gather_kernel(
    const float* __restrict__ z, const float* __restrict__ emb,
    const int* __restrict__ idx_ws, float* __restrict__ zq_out,
    float* __restrict__ row_loss)
{
    const int row = blockIdx.x;
    const int t   = threadIdx.x;
    const int k   = idx_ws[row];
    const float e  = emb[(size_t)k * D_MODEL + t];
    const float zv = z[(size_t)row * D_MODEL + t];
    zq_out[(size_t)row * D_MODEL + t] = e;   // straight-through == z_q numerically

    __shared__ float red[256];
    const float d = e - zv;
    red[t] = d * d;
    __syncthreads();
    #pragma unroll
    for (int s = 128; s > 0; s >>= 1) {
        if (t < s) red[t] += red[t + s];
        __syncthreads();
    }
    if (t == 0) row_loss[row] = red[0];
}

// ---------------------------------------------------------------------------
// Kernel 4: deterministic reduction of row losses -> scalar total loss.
// ---------------------------------------------------------------------------
__global__ __launch_bounds__(256) void vq_loss_kernel(
    const float* __restrict__ row_loss, float* __restrict__ loss_out)
{
    const int t = threadIdx.x;
    float acc = 0.0f;
    for (int i = t; i < N_ROWS; i += 256) acc += row_loss[i];
    __shared__ float red[256];
    red[t] = acc;
    __syncthreads();
    #pragma unroll
    for (int s = 128; s > 0; s >>= 1) {
        if (t < s) red[t] += red[t + s];
        __syncthreads();
    }
    if (t == 0) {
        const float mean = red[0] / (float)((size_t)N_ROWS * D_MODEL);
        *loss_out = mean + COMMIT_BETA * mean;  // codebook + beta*commitment
    }
}

extern "C" void kernel_launch(void* const* d_in, const int* in_sizes, int n_in,
                              void* d_out, int out_size, void* d_ws, size_t ws_size,
                              hipStream_t stream)
{
    const float* z   = (const float*)d_in[0];   // [32768, 256]
    const float* emb = (const float*)d_in[1];   // [2048, 256]

    // Workspace layout
    char*   ws       = (char*)d_ws;
    float*  e_norms  = (float*)ws;                                   // 2048 f32
    __bf16* emb_bf   = (__bf16*)(ws + 8192);                         // 1 MB
    int*    idx_ws   = (int*)(ws + 8192 + (size_t)NUM_EMB * D_MODEL * 2);
    float*  row_loss = (float*)((char*)idx_ws + (size_t)N_ROWS * 4);

    // Output layout: [z_q (N*D f32)] [loss (1 f32)] [indices-as-f32 (N)]
    float* out      = (float*)d_out;
    float* zq_out   = out;
    float* loss_out = out + (size_t)N_ROWS * D_MODEL;
    float* idx_out  = loss_out + 1;

    vq_prep_kernel  <<<NUM_EMB,        256, 0, stream>>>(emb, emb_bf, e_norms);
    vq_argmin_kernel<<<N_ROWS / 128,   256, 0, stream>>>(z, emb_bf, e_norms,
                                                         idx_ws, idx_out);
    vq_gather_kernel<<<N_ROWS,         256, 0, stream>>>(z, emb, idx_ws,
                                                         zq_out, row_loss);
    vq_loss_kernel  <<<1,              256, 0, stream>>>(row_loss, loss_out);
}
